// AttLayer_theano_87634512708218
// MI455X (gfx1250) — matrix operand order repaired and checked by reading, not compiled, not run on hardware
//
#include <hip/hip_runtime.h>

typedef __attribute__((ext_vector_type(16))) _Float16 v16h;
typedef __attribute__((ext_vector_type(8)))  float    v8f;
typedef __attribute__((ext_vector_type(4)))  float    f4;

#define B_      64
#define T_      4096
#define D_      256
#define A_      32
#define SPLITS  16
#define WAVES   2
#define ROWS_PER_BLOCK (T_ / SPLITS)            // 256
#define ROWS_PER_WAVE  (ROWS_PER_BLOCK / WAVES) // 128
#define TILES_PER_WAVE (ROWS_PER_WAVE / 16)     // 8
#define ROWF    260                              // padded row stride in floats (1040 B)
#define EPSI    1e-7f

// Issue async DMA of one 16x256 f32 tile (16 KB) into LDS (padded rows).
// GVS mode: saddr = batch base (SGPR pair), vaddr = per-lane byte offset,
// vdst = per-lane LDS byte address; offset: immediate applies to both sides.
__device__ __forceinline__ void async_copy_tile(const float* __restrict__ xb,
                                                int tRow0, unsigned tbase, int lane)
{
    unsigned goff = (unsigned)(tRow0 * 1024) + lane * 16;   // bytes within batch
    unsigned lofs = tbase + lane * 16;
#pragma unroll
    for (int r = 0; r < 16; ++r) {
        asm volatile("global_load_async_to_lds_b128 %0, %1, %2 offset:0"
                     :: "v"(lofs), "v"(goff), "s"(xb) : "memory");
        asm volatile("global_load_async_to_lds_b128 %0, %1, %2 offset:512"
                     :: "v"(lofs), "v"(goff), "s"(xb) : "memory");
        goff += 1024;    // global row stride
        lofs += 1040;    // padded LDS row stride (bank-conflict-free reads)
    }
}

// Pass 1: async-stage x tile -> LDS, WMMA f16 score GEMM, tanh/dot(u)/exp,
// weighted sum from LDS. One workgroup = one (batch, T-split); 2 waves.
__global__ __launch_bounds__(64) void att_pass1(
    const float* __restrict__ x, const unsigned char* __restrict__ mask,
    const float* __restrict__ W, const float* __restrict__ bias,
    const float* __restrict__ u, float* __restrict__ wsAcc, float* __restrict__ wsDen)
{
    __shared__ v16h  Wlds[8 * 2 * 32];                 // 16 KB, WMMA-A operand order
    __shared__ __align__(16) float tileF[WAVES][16 * ROWF]; // 2 x 16.25 KB
    __shared__ f4    accbuf[WAVES][64];                // 2 KB
    __shared__ float denbuf[WAVES];

    const int tid    = threadIdx.x;
    const int wave   = tid >> 5;
    const int lane   = tid & 31;
    const int laneLo = lane & 15;
    const int laneHi = lane >> 4;

    const int bIdx = blockIdx.x / SPLITS;
    const int sIdx = blockIdx.x % SPLITS;
    const float* xb = x + (size_t)bIdx * T_ * D_;       // batch base (uniform -> SGPRs)
    const int rowBase = sIdx * ROWS_PER_BLOCK + wave * ROWS_PER_WAVE;

    const unsigned tbase = (unsigned)(size_t)&tileF[wave][0];

    // kick off DMA of this wave's first tile before doing any setup work
    async_copy_tile(xb, rowBase, tbase, lane);

    // ---- cooperative W(f32,[D,A]) -> LDS f16 in exact A-operand lane order ----
    {
        _Float16* wl = (_Float16*)Wlds;
        for (int i = tid; i < 8 * 2 * 32 * 16; i += 64) {
            int e    = i & 15;
            int ln   = (i >> 4) & 31;
            int tile = (i >> 9) & 1;
            int kb   = (i >> 10);
            int a    = tile * 16 + (ln & 15);                             // A row M
            int d    = 32 * kb + 16 * (ln >> 4) + e;                      // K index
            wl[i] = (_Float16)W[d * A_ + a];
        }
    }
    float bias0[8], bias1[8], u0[8], u1[8];
#pragma unroll
    for (int r = 0; r < 8; ++r) {
        int a0 = r + 8 * laneHi;
        bias0[r] = bias[a0];      u0[r] = u[a0];
        bias1[r] = bias[a0 + 16]; u1[r] = u[a0 + 16];
    }
    __syncthreads();

    f4 accLo = {0.f, 0.f, 0.f, 0.f};
    f4 accHi = {0.f, 0.f, 0.f, 0.f};
    float dacc = 0.f;

    const float* rb = &tileF[wave][laneLo * ROWF + 16 * laneHi]; // B-operand view
    const float* tb = &tileF[wave][0];                           // row-major view

    for (int tile = 0; tile < TILES_PER_WAVE; ++tile) {
        const int tRow0 = rowBase + tile * 16;

        // wait for the staged tile
        asm volatile("s_wait_asynccnt 0" ::: "memory");

        // ---- score GEMM: D[a,t] = sum_d W[d,a]*x[t,d], 2 a-tiles, K=256 ----
        v8f c0 = {0.f,0.f,0.f,0.f,0.f,0.f,0.f,0.f};
        v8f c1 = {0.f,0.f,0.f,0.f,0.f,0.f,0.f,0.f};
#pragma unroll
        for (int kb = 0; kb < 8; ++kb) {
            f4 f0 = *(const f4*)(rb + 32 * kb + 0);
            f4 f1 = *(const f4*)(rb + 32 * kb + 4);
            f4 f2 = *(const f4*)(rb + 32 * kb + 8);
            f4 f3 = *(const f4*)(rb + 32 * kb + 12);
            v16h bOp;
#pragma unroll
            for (int e = 0; e < 4; ++e) {
                bOp[e]      = (_Float16)f0[e];
                bOp[4 + e]  = (_Float16)f1[e];
                bOp[8 + e]  = (_Float16)f2[e];
                bOp[12 + e] = (_Float16)f3[e];
            }
            v16h a0 = Wlds[(kb * 2 + 0) * 32 + lane];
            v16h a1 = Wlds[(kb * 2 + 1) * 32 + lane];
            c0 = __builtin_amdgcn_wmma_f32_16x16x32_f16(false, a0, false, bOp,
                                                        (short)0, c0, false, false);
            c1 = __builtin_amdgcn_wmma_f32_16x16x32_f16(false, a1, false, bOp,
                                                        (short)0, c1, false, false);
        }

        // ---- s[t] = sum_a tanh(c + b[a]) * u[a]; one xor16 completes the a-sum
        float s = 0.f;
#pragma unroll
        for (int r = 0; r < 8; ++r) {
            s += tanhf(c0[r] + bias0[r]) * u0[r];
            s += tanhf(c1[r] + bias1[r]) * u1[r];
        }
        s += __shfl_xor(s, 16, 32);                 // every lane: s for t = lane%16
        float ev = expf(s);
        ev *= (float)mask[(size_t)bIdx * T_ + tRow0 + laneLo];
        dacc += ev;                                  // halves duplicate; fixed below

        // ---- weighted sum from LDS: acc[d] += e[t]*x[t,d]; lane owns 4 d's (+128)
#pragma unroll
        for (int t = 0; t < 16; ++t) {
            float w  = __shfl(ev, t, 32);
            f4 xlo = *(const f4*)(tb + t * ROWF + 4 * lane);
            f4 xhi = *(const f4*)(tb + t * ROWF + 128 + 4 * lane);
            accLo += w * xlo;
            accHi += w * xhi;
        }

        // stage next tile (ds reads and async DMA writes are unordered: drain first)
        if (tile + 1 < TILES_PER_WAVE) {
            asm volatile("s_wait_dscnt 0" ::: "memory");
            async_copy_tile(xb, tRow0 + 16, tbase, lane);
        }
    }

    // wave denominator (each 16-lane group holds the same 16 e-sums)
    float dtot = dacc;
    dtot += __shfl_xor(dtot, 1, 32);
    dtot += __shfl_xor(dtot, 2, 32);
    dtot += __shfl_xor(dtot, 4, 32);
    dtot += __shfl_xor(dtot, 8, 32);
    if (lane == 0) denbuf[wave] = dtot;
    accbuf[wave][lane]      = accLo;     // d = 4*lane .. 4*lane+3
    accbuf[wave][32 + lane] = accHi;     // d = 128 + 4*lane ..
    __syncthreads();

    // block combine (deterministic), write split partials to workspace
    const float* ab = (const float*)accbuf;
#pragma unroll
    for (int dd = 0; dd < 4; ++dd) {
        int d = tid + dd * 64;
        wsAcc[(size_t)blockIdx.x * D_ + d] = ab[d] + ab[D_ + d];
    }
    if (tid == 0) wsDen[blockIdx.x] = denbuf[0] + denbuf[1];
}

// Pass 2: combine SPLITS partials per batch and normalize.
__global__ __launch_bounds__(256) void att_pass2(
    const float* __restrict__ wsAcc, const float* __restrict__ wsDen,
    float* __restrict__ out)
{
    const int b = blockIdx.x, d = threadIdx.x;
    float s = 0.f, den = 0.f;
#pragma unroll
    for (int sp = 0; sp < SPLITS; ++sp) {
        s   += wsAcc[(size_t)(b * SPLITS + sp) * D_ + d];
        den += wsDen[b * SPLITS + sp];
    }
    out[b * D_ + d] = s / (den + EPSI);
}

extern "C" void kernel_launch(void* const* d_in, const int* in_sizes, int n_in,
                              void* d_out, int out_size, void* d_ws, size_t ws_size,
                              hipStream_t stream) {
    const float*         x    = (const float*)d_in[0];
    const unsigned char* mask = (const unsigned char*)d_in[1];
    const float*         W    = (const float*)d_in[2];
    const float*         bias = (const float*)d_in[3];
    const float*         u    = (const float*)d_in[4];

    float* wsAcc = (float*)d_ws;                          // [B*SPLITS, 256]
    float* wsDen = wsAcc + (size_t)B_ * SPLITS * D_;      // [B*SPLITS]

    att_pass1<<<B_ * SPLITS, 64, 0, stream>>>(x, mask, W, bias, u, wsAcc, wsDen);
    att_pass2<<<B_, 256, 0, stream>>>(wsAcc, wsDen, (float*)d_out);
}
